// PartitionAttentionLayer_25031069401124
// MI455X (gfx1250) — matrix operand order repaired and checked by reading, not compiled, not run on hardware
//
#include <hip/hip_runtime.h>
#include <hip/hip_bf16.h>
#include <stdint.h>

// ---------------------------------------------------------------------------
// Types
// ---------------------------------------------------------------------------
typedef __bf16 bf16;
typedef __attribute__((ext_vector_type(16))) __bf16 bfx16;
typedef __attribute__((ext_vector_type(8)))  __bf16 bfx8;
typedef __attribute__((ext_vector_type(8)))  float  fx8;

#define M_TOK   100352L   // 32 * 64 * 49 tokens
#define M_PAD   100416L   // + 64 pad rows for window over-read
#define CCH     256
#define NQKV    768
#define NH      8
#define HD      32
#define P2W     49        // tokens per window
#define NWIN    2048      // 32 * 64 windows

// ---------------------------------------------------------------------------
// WMMA helpers.  A-operand (16x32 bf16): lane group (lane>>4) selects k-half;
// per lane contiguous 16 halves at k-chunk (lane>>4)*16 (matches B).
// B-operand (32x16 bf16): lane holds column n=lane&15, contiguous 16 halves
// at k-chunk (lane>>4)*16.
// ---------------------------------------------------------------------------
__device__ __forceinline__ bfx16 load_a16(const bf16* p) {
    bfx8 lo = *(const bfx8*)p;
    bfx8 hi = *(const bfx8*)(p + 8);
    return __builtin_shufflevector(lo, hi, 0,1,2,3,4,5,6,7,8,9,10,11,12,13,14,15);
}

__device__ __forceinline__ fx8 wmma_bf16(bfx16 a, bfx16 b, fx8 c) {
    return __builtin_amdgcn_wmma_f32_16x16x32_bf16(
        /*neg_a=*/false, a, /*neg_b=*/false, b,
        /*c_mod=*/(short)0, c, /*reuse_a=*/false, /*reuse_b=*/false);
}

// Async global -> LDS copy of 16 bytes (ASYNCcnt-tracked, CDNA5 TDM-lite path)
__device__ __forceinline__ void async_copy_b128(unsigned lds_addr, const void* gaddr) {
    asm volatile("global_load_async_to_lds_b128 %0, %1, off"
                 :: "v"(lds_addr), "v"((unsigned long long)(uintptr_t)gaddr)
                 : "memory");
}
__device__ __forceinline__ void wait_asynccnt0() {
    asm volatile("s_wait_asynccnt 0x0" ::: "memory");
}

// ---------------------------------------------------------------------------
// Cooperative GEMM mainloop.
// Block = 256 threads = 8 waves arranged 2(M) x 4(N): macro-tile 32 x 256.
// A stripe (32 rows) double-buffered in LDS, streamed in 64-wide K chunks via
// global_load_async_to_lds_b128; every wave computes a 16x64 register tile,
// reading A fragments from LDS (ds_load) and B (L2-resident weights) from
// global.  acc[4] laid out per ISA C/D layout.
// ---------------------------------------------------------------------------
__device__ __forceinline__ void wmma_tile_coop(
    const bf16* __restrict__ A, const bf16* __restrict__ Wt, int K,
    long row0, long col0, bf16* sA /* [2][32*64] */, fx8 acc[4])
{
    const int tid  = threadIdx.x;
    const int lane = tid & 31;
    const int wid  = tid >> 5;
    const int rg   = wid >> 2;          // row-group 0..1
    const int cg   = wid & 3;           // col-group 0..3
    const int kh   = (lane >> 4) * 16;  // A sub-chunk base
    const int kb   = (lane >> 4) * 16;  // B chunk base
    const int srow = tid >> 3;          // staging row 0..31
    const int sseg = tid & 7;           // staging 8-half segment
    const int arow = rg * 16 + (lane & 15);
    const long wn0 = col0 + cg * 64 + (lane & 15);
    const int nk   = K / 64;

    // stage k-chunk 0 into buffer 0
    async_copy_b128((unsigned)(uintptr_t)&sA[srow * 64 + sseg * 8],
                    A + (row0 + srow) * (long)K + sseg * 8);

    for (int kc = 0; kc < nk; ++kc) {
        wait_asynccnt0();
        __syncthreads();                 // buf[kc&1] ready everywhere
        if (kc + 1 < nk) {               // prefetch next chunk into other buffer
            async_copy_b128(
                (unsigned)(uintptr_t)&sA[((kc + 1) & 1) * 2048 + srow * 64 + sseg * 8],
                A + (row0 + srow) * (long)K + (kc + 1) * 64 + sseg * 8);
        }
        const bf16* ab = &sA[(kc & 1) * 2048 + arow * 64];
#pragma unroll
        for (int ks = 0; ks < 64; ks += 32) {
            bfx8 lo = *(const bfx8*)(ab + ks + kh);
            bfx8 hi = *(const bfx8*)(ab + ks + kh + 8);
            bfx16 a = __builtin_shufflevector(lo, hi,
                0,1,2,3,4,5,6,7,8,9,10,11,12,13,14,15);
            const int k0 = kc * 64 + ks;
#pragma unroll
            for (int nt = 0; nt < 4; ++nt) {
                bfx16 b = *(const bfx16*)(Wt + (wn0 + nt * 16) * (long)K + k0 + kb);
                acc[nt] = wmma_bf16(a, b, acc[nt]);
            }
        }
        __syncthreads();                 // done reading buf[kc&1] before overwrite
    }
}

// ---------------------------------------------------------------------------
// Weight transpose+convert: src f32 [K][N] -> dst bf16 [N][K]
// ---------------------------------------------------------------------------
__global__ void wconv_kernel(const float* __restrict__ src, bf16* __restrict__ dst,
                             int K, int N) {
    long idx = (long)blockIdx.x * blockDim.x + threadIdx.x;
    if (idx >= (long)K * N) return;
    int k = (int)(idx / N), n = (int)(idx % N);
    dst[(long)n * K + k] = (bf16)src[idx];
}

// pos_bias[(2P-1)^2, NH] gathered through rel_idx -> biasT[NH][49][49] f32
__global__ void bias_kernel(const float* __restrict__ pos_bias,
                            const int* __restrict__ rel_idx,
                            float* __restrict__ biasT) {
    int idx = blockIdx.x * blockDim.x + threadIdx.x;
    if (idx >= NH * P2W * P2W) return;
    int hd = idx / (P2W * P2W), r = idx % (P2W * P2W);
    biasT[idx] = pos_bias[rel_idx[r] * NH + hd];
}

// ---------------------------------------------------------------------------
// LN1 + window partition: x[B,C,56,56] -> h bf16 [M_TOK, 256]
// ---------------------------------------------------------------------------
__global__ __launch_bounds__(256)
void ln1_kernel(const float* __restrict__ x, const float* __restrict__ w,
                const float* __restrict__ b, bf16* __restrict__ h) {
    const int lane = threadIdx.x & 31;
    const long t = (long)blockIdx.x * (blockDim.x >> 5) + (threadIdx.x >> 5);
    if (t >= M_TOK) return;
    int bb = (int)(t / 3136), rr = (int)(t % 3136);
    int g = rr / P2W, p2 = rr % P2W;
    int hh = (g >> 3) * 7 + p2 / 7;
    int ww = (g & 7) * 7 + p2 % 7;
    const float* xb = x + (long)bb * CCH * 3136 + (long)hh * 56 + ww;
    float vals[8];
    float s = 0.f;
#pragma unroll
    for (int j = 0; j < 8; ++j) {
        int c = lane + j * 32;
        vals[j] = xb[(long)c * 3136];
        s += vals[j];
    }
#pragma unroll
    for (int o = 16; o > 0; o >>= 1) s += __shfl_xor(s, o, 32);
    float mean = s * (1.f / CCH);
    float vs = 0.f;
#pragma unroll
    for (int j = 0; j < 8; ++j) { float d = vals[j] - mean; vs += d * d; }
#pragma unroll
    for (int o = 16; o > 0; o >>= 1) vs += __shfl_xor(vs, o, 32);
    float inv = rsqrtf(vs * (1.f / CCH) + 1e-5f);
    bf16* hp = h + t * CCH;
#pragma unroll
    for (int j = 0; j < 8; ++j) {
        int c = lane + j * 32;
        hp[c] = (bf16)((vals[j] - mean) * inv * w[c] + b[c]);
    }
}

// LN2 over xp f32 [M_TOK,256] (contiguous rows) -> hm bf16
__global__ __launch_bounds__(256)
void ln2_kernel(const float* __restrict__ xp, const float* __restrict__ w,
                const float* __restrict__ b, bf16* __restrict__ hm) {
    const int lane = threadIdx.x & 31;
    const long t = (long)blockIdx.x * (blockDim.x >> 5) + (threadIdx.x >> 5);
    if (t >= M_TOK) return;
    const float* xr = xp + t * CCH;
    float vals[8];
    float s = 0.f;
#pragma unroll
    for (int j = 0; j < 8; ++j) { vals[j] = xr[lane + j * 32]; s += vals[j]; }
#pragma unroll
    for (int o = 16; o > 0; o >>= 1) s += __shfl_xor(s, o, 32);
    float mean = s * (1.f / CCH);
    float vs = 0.f;
#pragma unroll
    for (int j = 0; j < 8; ++j) { float d = vals[j] - mean; vs += d * d; }
#pragma unroll
    for (int o = 16; o > 0; o >>= 1) vs += __shfl_xor(vs, o, 32);
    float inv = rsqrtf(vs * (1.f / CCH) + 1e-5f);
    bf16* hp = hm + t * CCH;
#pragma unroll
    for (int j = 0; j < 8; ++j) {
        int c = lane + j * 32;
        hp[c] = (bf16)((vals[j] - mean) * inv * w[c] + b[c]);
    }
}

// ---------------------------------------------------------------------------
// GEMM 1: qkv = h @ qkv_w + b ; k-part scaled by 256^-0.5 ; bf16 out [M_PAD,768]
// ---------------------------------------------------------------------------
__global__ __launch_bounds__(256)
void gemm_qkv_kernel(const bf16* __restrict__ h, const bf16* __restrict__ wT,
                     const float* __restrict__ bias, bf16* __restrict__ qkv) {
    __shared__ __align__(32) bf16 sA[2 * 32 * 64];
    const int nbn = NQKV / 256;
    long row0 = (long)(blockIdx.x / nbn) * 32;
    long col0 = (long)(blockIdx.x % nbn) * 256;
    fx8 acc[4] = {};
    wmma_tile_coop(h, wT, CCH, row0, col0, sA, acc);
    const int lane = threadIdx.x & 31;
    const int wid  = threadIdx.x >> 5;
    long rw = row0 + (wid >> 2) * 16;
    long cw = col0 + (wid & 3) * 64;
#pragma unroll
    for (int nt = 0; nt < 4; ++nt)
#pragma unroll
        for (int r = 0; r < 8; ++r) {
            long m = rw + r + ((lane >> 4) << 3);
            long n = cw + nt * 16 + (lane & 15);
            float v = acc[nt][r] + bias[n];
            if (n >= 256 && n < 512) v *= 0.0625f;   // 256^-0.5
            qkv[m * NQKV + n] = (bf16)v;
        }
}

// ---------------------------------------------------------------------------
// Attention per (window, head): 128 threads = 4 waves.
// ---------------------------------------------------------------------------
__global__ __launch_bounds__(128)
void attn_kernel(const bf16* __restrict__ qkv, const float* __restrict__ biasT,
                 bf16* __restrict__ o) {
    const int wh = blockIdx.x;       // window*8 + head
    const int w  = wh >> 3;
    const int hd = wh & 7;
    const int tid = threadIdx.x;
    const int lane = tid & 31;
    const int wv = tid >> 5;
    __shared__ __align__(32) float s_sc[64 * 64];
    __shared__ __align__(32) bf16  s_at[64 * 64];
    const long base = (long)w * P2W * NQKV;
    const int kh = (lane >> 4) * 16;
    const int kb = (lane >> 4) * 16;

    // ---- scores = q @ k^T (K=32, one wmma per n-tile) ----
    {
        int i = wv * 16 + (lane & 15);
        const bf16* qp = qkv + base + (long)i * NQKV + hd * HD;
        bfx16 a = load_a16(qp + kh);
        fx8 acc[4] = {};
#pragma unroll
        for (int nt = 0; nt < 4; ++nt) {
            int j = nt * 16 + (lane & 15);
            const bf16* kp = qkv + base + (long)j * NQKV + 256 + hd * HD + kb;
            bfx16 b = *(const bfx16*)kp;
            acc[nt] = wmma_bf16(a, b, acc[nt]);
        }
#pragma unroll
        for (int nt = 0; nt < 4; ++nt)
#pragma unroll
            for (int r = 0; r < 8; ++r) {
                int ii = wv * 16 + r + ((lane >> 4) << 3);
                int jj = nt * 16 + (lane & 15);
                s_sc[ii * 64 + jj] = acc[nt][r];
            }
    }
    __syncthreads();

    // ---- softmax over j<49 with relative-position bias ----
    if (tid < 64) {
        int i = tid;
        if (i < P2W) {
            const float* bp = biasT + hd * (P2W * P2W) + i * P2W;
            float mx = -1e30f;
            for (int j = 0; j < P2W; ++j) {
                float v = s_sc[i * 64 + j] + bp[j];
                mx = fmaxf(mx, v);
            }
            float sum = 0.f;
            for (int j = 0; j < P2W; ++j) {
                float e = __expf(s_sc[i * 64 + j] + bp[j] - mx);
                s_sc[i * 64 + j] = e;
                sum += e;
            }
            float rs = 1.f / sum;
            for (int j = 0; j < P2W; ++j) s_at[i * 64 + j] = (bf16)(s_sc[i * 64 + j] * rs);
            for (int j = P2W; j < 64; ++j) s_at[i * 64 + j] = (bf16)0.f;
        } else {
            for (int j = 0; j < 64; ++j) s_at[i * 64 + j] = (bf16)0.f;
        }
    }
    __syncthreads();

    // ---- o = attn @ v (K=64 -> 2 k-steps, N=32 -> 2 n-tiles) ----
    {
        int i = wv * 16 + (lane & 15);
        fx8 acc[2] = {};
#pragma unroll
        for (int kk = 0; kk < 2; ++kk) {
            const bf16* ar = &s_at[i * 64 + kk * 32 + kh];
            bfx8 lo = *(const bfx8*)ar;
            bfx8 hi = *(const bfx8*)(ar + 8);
            bfx16 a = __builtin_shufflevector(lo, hi,
                0,1,2,3,4,5,6,7,8,9,10,11,12,13,14,15);
#pragma unroll
            for (int nt = 0; nt < 2; ++nt) {
                int d = nt * 16 + (lane & 15);
                const bf16* vp = qkv + base + 512 + hd * HD + d;
                bfx16 bm;
#pragma unroll
                for (int e = 0; e < 16; ++e) {
                    int j = kk * 32 + kb + e;          // token index (padded reads OK)
                    bm[e] = vp[(long)j * NQKV];
                }
                acc[nt] = wmma_bf16(a, bm, acc[nt]);
            }
        }
#pragma unroll
        for (int nt = 0; nt < 2; ++nt)
#pragma unroll
            for (int r = 0; r < 8; ++r) {
                int ii = wv * 16 + r + ((lane >> 4) << 3);
                if (ii < P2W) {
                    int d = nt * 16 + (lane & 15);
                    o[((long)w * P2W + ii) * CCH + hd * HD + d] = (bf16)acc[nt][r];
                }
            }
    }
}

// ---------------------------------------------------------------------------
// GEMM 2: xp = (o @ merge_w + merge_b) * attn_scale   (f32 out)
// ---------------------------------------------------------------------------
__global__ __launch_bounds__(256)
void gemm_merge_kernel(const bf16* __restrict__ o, const bf16* __restrict__ wT,
                       const float* __restrict__ bias, const float* __restrict__ scale,
                       float* __restrict__ xp) {
    __shared__ __align__(32) bf16 sA[2 * 32 * 64];
    long row0 = (long)blockIdx.x * 32;   // N = 256 -> single col block
    long col0 = 0;
    fx8 acc[4] = {};
    wmma_tile_coop(o, wT, CCH, row0, col0, sA, acc);
    const int lane = threadIdx.x & 31;
    const int wid  = threadIdx.x >> 5;
    long rw = row0 + (wid >> 2) * 16;
    long cw = col0 + (wid & 3) * 64;
#pragma unroll
    for (int nt = 0; nt < 4; ++nt)
#pragma unroll
        for (int r = 0; r < 8; ++r) {
            long m = rw + r + ((lane >> 4) << 3);
            long n = cw + nt * 16 + (lane & 15);
            xp[m * CCH + n] = (acc[nt][r] + bias[n]) * scale[n];
        }
}

// ---------------------------------------------------------------------------
// GEMM 3: t1 = gelu(hm @ w1 + b1)  (exact gelu, bf16 out [M_TOK,1024])
// ---------------------------------------------------------------------------
__global__ __launch_bounds__(256)
void gemm_mlp1_kernel(const bf16* __restrict__ hm, const bf16* __restrict__ wT,
                      const float* __restrict__ bias, bf16* __restrict__ t1) {
    __shared__ __align__(32) bf16 sA[2 * 32 * 64];
    const int nbn = 1024 / 256;
    long row0 = (long)(blockIdx.x / nbn) * 32;
    long col0 = (long)(blockIdx.x % nbn) * 256;
    fx8 acc[4] = {};
    wmma_tile_coop(hm, wT, CCH, row0, col0, sA, acc);
    const int lane = threadIdx.x & 31;
    const int wid  = threadIdx.x >> 5;
    long rw = row0 + (wid >> 2) * 16;
    long cw = col0 + (wid & 3) * 64;
#pragma unroll
    for (int nt = 0; nt < 4; ++nt)
#pragma unroll
        for (int r = 0; r < 8; ++r) {
            long m = rw + r + ((lane >> 4) << 3);
            long n = cw + nt * 16 + (lane & 15);
            float v = acc[nt][r] + bias[n];
            v = 0.5f * v * (1.f + erff(v * 0.70710678f));
            t1[m * 1024 + n] = (bf16)v;
        }
}

// ---------------------------------------------------------------------------
// GEMM 4: y = (xp + t1 @ w2 + b2) * mlp_scale, scattered to [B,C,56,56]
// ---------------------------------------------------------------------------
__global__ __launch_bounds__(256)
void gemm_mlp2_kernel(const bf16* __restrict__ t1, const bf16* __restrict__ wT,
                      const float* __restrict__ bias, const float* __restrict__ scale,
                      const float* __restrict__ xp, float* __restrict__ y) {
    __shared__ __align__(32) bf16 sA[2 * 32 * 64];
    long row0 = (long)blockIdx.x * 32;   // N = 256 -> single col block
    long col0 = 0;
    fx8 acc[4] = {};
    wmma_tile_coop(t1, wT, 1024, row0, col0, sA, acc);
    const int lane = threadIdx.x & 31;
    const int wid  = threadIdx.x >> 5;
    long rw = row0 + (wid >> 2) * 16;
    long cw = col0 + (wid & 3) * 64;
#pragma unroll
    for (int nt = 0; nt < 4; ++nt)
#pragma unroll
        for (int r = 0; r < 8; ++r) {
            long m = rw + r + ((lane >> 4) << 3);
            long n = cw + nt * 16 + (lane & 15);
            float v = (xp[m * CCH + n] + acc[nt][r] + bias[n]) * scale[n];
            // departition: token m -> (b, hh, ww)
            int bb = (int)(m / 3136), rr = (int)(m % 3136);
            int g = rr / P2W, p2 = rr % P2W;
            int hh = (g >> 3) * 7 + p2 / 7;
            int ww = (g & 7) * 7 + p2 % 7;
            y[(((long)bb * CCH + n) * 56 + hh) * 56 + ww] = v;
        }
}

// ---------------------------------------------------------------------------
// Host-side orchestration
// ---------------------------------------------------------------------------
extern "C" void kernel_launch(void* const* d_in, const int* in_sizes, int n_in,
                              void* d_out, int out_size, void* d_ws, size_t ws_size,
                              hipStream_t stream) {
    (void)in_sizes; (void)n_in; (void)out_size; (void)ws_size;
    const float* x        = (const float*)d_in[0];
    const float* ln1_w    = (const float*)d_in[1];
    const float* ln1_b    = (const float*)d_in[2];
    const float* qkv_w    = (const float*)d_in[3];
    const float* qkv_b    = (const float*)d_in[4];
    const float* merge_w  = (const float*)d_in[5];
    const float* merge_b  = (const float*)d_in[6];
    const float* pos_bias = (const float*)d_in[7];
    const float* attn_sc  = (const float*)d_in[8];
    const float* ln2_w    = (const float*)d_in[9];
    const float* ln2_b    = (const float*)d_in[10];
    const float* mlp_w1   = (const float*)d_in[11];
    const float* mlp_b1   = (const float*)d_in[12];
    const float* mlp_w2   = (const float*)d_in[13];
    const float* mlp_b2   = (const float*)d_in[14];
    const float* mlp_sc   = (const float*)d_in[15];
    const int*   rel_idx  = (const int*)d_in[16];
    float* y = (float*)d_out;

    char* p = (char*)d_ws;
    auto take = [&](size_t bytes) -> char* {
        char* r = p;
        p += (bytes + 255) & ~(size_t)255;
        return r;
    };
    bf16*  wqkvT  = (bf16*)take((size_t)NQKV * CCH * 2);
    bf16*  wmrgT  = (bf16*)take((size_t)CCH * CCH * 2);
    bf16*  w1T    = (bf16*)take((size_t)1024 * CCH * 2);
    bf16*  w2T    = (bf16*)take((size_t)CCH * 1024 * 2);
    float* biasT  = (float*)take((size_t)NH * P2W * P2W * 4);
    bf16*  h      = (bf16*)take((size_t)M_TOK * CCH * 2);
    bf16*  qkv    = (bf16*)take((size_t)M_PAD * NQKV * 2);
    bf16*  obuf   = (bf16*)take((size_t)M_TOK * CCH * 2);
    float* xp     = (float*)take((size_t)M_TOK * CCH * 4);
    bf16*  hm     = (bf16*)take((size_t)M_TOK * CCH * 2);
    bf16*  t1     = (bf16*)take((size_t)M_TOK * 1024 * 2);

    // weight transposes + bias gather
    wconv_kernel<<<(CCH * NQKV + 255) / 256, 256, 0, stream>>>(qkv_w, wqkvT, CCH, NQKV);
    wconv_kernel<<<(CCH * CCH + 255) / 256, 256, 0, stream>>>(merge_w, wmrgT, CCH, CCH);
    wconv_kernel<<<(CCH * 1024 + 255) / 256, 256, 0, stream>>>(mlp_w1, w1T, CCH, 1024);
    wconv_kernel<<<(1024 * CCH + 255) / 256, 256, 0, stream>>>(mlp_w2, w2T, 1024, CCH);
    bias_kernel<<<(NH * P2W * P2W + 255) / 256, 256, 0, stream>>>(pos_bias, rel_idx, biasT);

    // LN1 + partition: 8 tokens per 256-thread block
    ln1_kernel<<<(int)(M_TOK / 8), 256, 0, stream>>>(x, ln1_w, ln1_b, h);

    // zero qkv pad tail so over-read windows see finite zeros
    hipMemsetAsync(qkv + (size_t)M_TOK * NQKV, 0, (size_t)(M_PAD - M_TOK) * NQKV * 2, stream);

    // QKV GEMM: (M/32) x (768/256) macro-tiles
    gemm_qkv_kernel<<<(int)((M_TOK / 32) * (NQKV / 256)), 256, 0, stream>>>(h, wqkvT, qkv_b, qkv);

    // attention: one block per (window, head)
    attn_kernel<<<NWIN * NH, 128, 0, stream>>>(qkv, biasT, obuf);

    // merge GEMM: (M/32) x 1 macro-tiles
    gemm_merge_kernel<<<(int)(M_TOK / 32), 256, 0, stream>>>(obuf, wmrgT, merge_b, attn_sc, xp);

    // LN2
    ln2_kernel<<<(int)(M_TOK / 8), 256, 0, stream>>>(xp, ln2_w, ln2_b, hm);

    // MLP GEMMs
    gemm_mlp1_kernel<<<(int)((M_TOK / 32) * (1024 / 256)), 256, 0, stream>>>(hm, w1T, mlp_b1, t1);
    gemm_mlp2_kernel<<<(int)(M_TOK / 32), 256, 0, stream>>>(t1, w2T, mlp_b2, mlp_sc, xp, y);
}
